// GCN_87780541596204
// MI455X (gfx1250) — matrix operand order repaired and checked by reading, not compiled, not run on hardware
//
#include <hip/hip_runtime.h>
#include <hip/hip_bf16.h>

typedef __bf16 bf16_t;
typedef __attribute__((ext_vector_type(16))) __bf16 v16bf;
typedef __attribute__((ext_vector_type(8)))  __bf16 v8bf;
typedef __attribute__((ext_vector_type(8)))  float  v8f;
typedef __attribute__((ext_vector_type(4)))  float  v4f;

#define N_NODES 50000
#define N_EDGES 800000
#define IN_DIM  512
#define HID_DIM 64
#define OUT_DIM 40

// ---------- helpers ----------
// native f32 -> bf16 convert (gfx1250 has hardware bf16 cvt; lowers to v_cvt_*bf16_f32)
__device__ __forceinline__ bf16_t f2bf(float f) { return (bf16_t)f; }

// load 16 bf16 as two 16B chunks: [p, p+8) and [p+gap, p+gap+8)
__device__ __forceinline__ v16bf load2x8(const bf16_t* p, int gap) {
    v8bf lo = *(const v8bf*)(p);
    v8bf hi = *(const v8bf*)(p + gap);
    v16bf r;
#pragma unroll
    for (int i = 0; i < 8; ++i) { r[i] = lo[i]; r[8 + i] = hi[i]; }
    return r;
}

// ---------- prep kernels ----------
__global__ void k_deg(const int* __restrict__ dst, float* __restrict__ deg, int E) {
    int e = blockIdx.x * blockDim.x + threadIdx.x;
    if (e < E) atomicAdd(&deg[dst[e]], 1.0f);
}

__global__ void k_dis(float* __restrict__ deg, int n) {
    int i = blockIdx.x * blockDim.x + threadIdx.x;
    if (i < n) deg[i] = rsqrtf(deg[i] + 1.0f);     // +1 = self loop; always > 0
}

// W1 [512x64] f32 -> W1t [64x512] bf16
__global__ void k_w1t(const float* __restrict__ W1, bf16_t* __restrict__ W1t) {
    int idx = blockIdx.x * blockDim.x + threadIdx.x;
    if (idx >= IN_DIM * HID_DIM) return;
    int k = idx >> 6, n = idx & 63;
    W1t[n * IN_DIM + k] = f2bf(W1[idx]);
}

// W2 [64x40] f32 -> W2t [64x64] bf16, zero padded cols 40..63
__global__ void k_w2t(const float* __restrict__ W2, bf16_t* __restrict__ W2t) {
    int idx = blockIdx.x * blockDim.x + threadIdx.x;
    if (idx >= 64 * 64) return;
    int k = idx >> 6, n = idx & 63;
    W2t[n * 64 + k] = (n < OUT_DIM) ? f2bf(W2[k * OUT_DIM + n]) : f2bf(0.0f);
}

// ---------- WMMA GEMMs: C[M x 64] = A[M x K] * Bt[64 x K]^T ----------
// one wave handles one 16-row tile across all four 16-col tiles
__global__ void __launch_bounds__(256) k_gemm_f32a(
    const float* __restrict__ A, const bf16_t* __restrict__ Bt,
    float* __restrict__ C, int M, int K)
{
    const int lane  = threadIdx.x & 31;
    const int wave  = threadIdx.x >> 5;
    const int tile  = blockIdx.x * 8 + wave;
    if (tile * 16 >= M) return;

    const int nlane = lane & 15;
    const int half8 = (lane < 16) ? 0 : 8;   // A K-chunk base & D row offset
    const int bk    = (lane < 16) ? 0 : 8;   // B K-chunk base

    const float*  arow = A + (size_t)(tile * 16 + nlane) * K;
    const bf16_t* b0 = Bt + (size_t)(nlane +  0) * K + bk;
    const bf16_t* b1 = Bt + (size_t)(nlane + 16) * K + bk;
    const bf16_t* b2 = Bt + (size_t)(nlane + 32) * K + bk;
    const bf16_t* b3 = Bt + (size_t)(nlane + 48) * K + bk;

    v8f acc0 = {0,0,0,0,0,0,0,0}, acc1 = acc0, acc2 = acc0, acc3 = acc0;

    for (int k0 = 0; k0 < K; k0 += 32) {
        // prefetch A two K-iterations ahead (gfx1250 global_prefetch_b8 path)
        if (k0 + 64 < K) __builtin_prefetch(arow + k0 + 64, 0, 1);

        // A fragment: f32 -> bf16 native conversion
        v4f f0 = *(const v4f*)(arow + k0 + half8);
        v4f f1 = *(const v4f*)(arow + k0 + half8 + 4);
        v4f f2 = *(const v4f*)(arow + k0 + half8 + 16);
        v4f f3 = *(const v4f*)(arow + k0 + half8 + 20);
        v16bf a;
#pragma unroll
        for (int i = 0; i < 4; ++i) {
            a[i]      = f2bf(f0[i]);
            a[4 + i]  = f2bf(f1[i]);
            a[8 + i]  = f2bf(f2[i]);
            a[12 + i] = f2bf(f3[i]);
        }
        acc0 = __builtin_amdgcn_wmma_f32_16x16x32_bf16(false, a, false, load2x8(b0 + k0, 16), (short)0, acc0, false, false);
        acc1 = __builtin_amdgcn_wmma_f32_16x16x32_bf16(false, a, false, load2x8(b1 + k0, 16), (short)0, acc1, false, false);
        acc2 = __builtin_amdgcn_wmma_f32_16x16x32_bf16(false, a, false, load2x8(b2 + k0, 16), (short)0, acc2, false, false);
        acc3 = __builtin_amdgcn_wmma_f32_16x16x32_bf16(false, a, false, load2x8(b3 + k0, 16), (short)0, acc3, false, false);
    }

    float* crow = C + (size_t)(tile * 16 + half8) * 64 + nlane;
#pragma unroll
    for (int r = 0; r < 8; ++r) {
        crow[(size_t)r * 64 +  0] = acc0[r];
        crow[(size_t)r * 64 + 16] = acc1[r];
        crow[(size_t)r * 64 + 32] = acc2[r];
        crow[(size_t)r * 64 + 48] = acc3[r];
    }
}

__global__ void __launch_bounds__(256) k_gemm_bf16a(
    const bf16_t* __restrict__ A, const bf16_t* __restrict__ Bt,
    float* __restrict__ C, int M, int K)
{
    const int lane  = threadIdx.x & 31;
    const int wave  = threadIdx.x >> 5;
    const int tile  = blockIdx.x * 8 + wave;
    if (tile * 16 >= M) return;

    const int nlane = lane & 15;
    const int half8 = (lane < 16) ? 0 : 8;
    const int bk    = (lane < 16) ? 0 : 8;

    const bf16_t* arow = A + (size_t)(tile * 16 + nlane) * K;
    const bf16_t* b0 = Bt + (size_t)(nlane +  0) * K + bk;
    const bf16_t* b1 = Bt + (size_t)(nlane + 16) * K + bk;
    const bf16_t* b2 = Bt + (size_t)(nlane + 32) * K + bk;
    const bf16_t* b3 = Bt + (size_t)(nlane + 48) * K + bk;

    v8f acc0 = {0,0,0,0,0,0,0,0}, acc1 = acc0, acc2 = acc0, acc3 = acc0;

    for (int k0 = 0; k0 < K; k0 += 32) {
        v16bf a = load2x8(arow + k0 + half8, 16);
        acc0 = __builtin_amdgcn_wmma_f32_16x16x32_bf16(false, a, false, load2x8(b0 + k0, 16), (short)0, acc0, false, false);
        acc1 = __builtin_amdgcn_wmma_f32_16x16x32_bf16(false, a, false, load2x8(b1 + k0, 16), (short)0, acc1, false, false);
        acc2 = __builtin_amdgcn_wmma_f32_16x16x32_bf16(false, a, false, load2x8(b2 + k0, 16), (short)0, acc2, false, false);
        acc3 = __builtin_amdgcn_wmma_f32_16x16x32_bf16(false, a, false, load2x8(b3 + k0, 16), (short)0, acc3, false, false);
    }

    float* crow = C + (size_t)(tile * 16 + half8) * 64 + nlane;
#pragma unroll
    for (int r = 0; r < 8; ++r) {
        crow[(size_t)r * 64 +  0] = acc0[r];
        crow[(size_t)r * 64 + 16] = acc1[r];
        crow[(size_t)r * 64 + 32] = acc2[r];
        crow[(size_t)r * 64 + 48] = acc3[r];
    }
}

// ---------- edge scatter: agg[dst,f] += dis[src]*dis[dst]*ht[src,f] ----------
__global__ void k_scatter(const int* __restrict__ src, const int* __restrict__ dst,
                          const float* __restrict__ dis, const float* __restrict__ ht,
                          float* __restrict__ agg, int E, int F, int ldht)
{
    long long idx = (long long)blockIdx.x * blockDim.x + threadIdx.x;
    if (idx >= (long long)E * F) return;
    int f = (int)(idx % F);
    int e = (int)(idx / F);
    int s = src[e], d = dst[e];
    float w = dis[s] * dis[d];
    atomicAdd(&agg[(size_t)d * F + f], w * ht[(size_t)s * ldht + f]);
}

// ---------- post layer 1: self-loop + bias + relu + dropout -> bf16 ----------
__global__ void k_post1(const float* __restrict__ agg1, const float* __restrict__ ht1,
                        const float* __restrict__ dis, const float* __restrict__ b1,
                        bf16_t* __restrict__ h1bf, int n)
{
    int idx = blockIdx.x * blockDim.x + threadIdx.x;
    if (idx >= n * HID_DIM) return;
    int i = idx >> 6, f = idx & 63;
    float di = dis[i];
    float v = agg1[idx] + di * di * ht1[idx] + b1[f];
    v = fmaxf(v, 0.0f);
    unsigned h = (unsigned)idx ^ 0x9E3779B9u;
    h ^= h >> 16; h *= 0x7feb352du; h ^= h >> 15; h *= 0x846ca68bu; h ^= h >> 16;
    v = (h & 1u) ? v * 2.0f : 0.0f;                 // p = 0.5, scale 1/(1-p)
    h1bf[idx] = f2bf(v);
}

// ---------- post layer 2: self-loop + bias + log_softmax ----------
__global__ void k_post2(const float* __restrict__ agg2, const float* __restrict__ ht2,
                        const float* __restrict__ dis, const float* __restrict__ b2,
                        float* __restrict__ out, int n)
{
    int i = blockIdx.x * blockDim.x + threadIdx.x;
    if (i >= n) return;
    float di2 = dis[i] * dis[i];
    float v[OUT_DIM];
    float mx = -3.402823466e38f;
#pragma unroll
    for (int f = 0; f < OUT_DIM; ++f) {
        float t = agg2[(size_t)i * OUT_DIM + f] + di2 * ht2[(size_t)i * 64 + f] + b2[f];
        v[f] = t; mx = fmaxf(mx, t);
    }
    float s = 0.0f;
#pragma unroll
    for (int f = 0; f < OUT_DIM; ++f) s += __expf(v[f] - mx);
    float ls = __logf(s);
#pragma unroll
    for (int f = 0; f < OUT_DIM; ++f) out[(size_t)i * OUT_DIM + f] = v[f] - mx - ls;
}

// ---------- launch ----------
extern "C" void kernel_launch(void* const* d_in, const int* in_sizes, int n_in,
                              void* d_out, int out_size, void* d_ws, size_t ws_size,
                              hipStream_t stream)
{
    const float* x   = (const float*)d_in[0];
    const int*   ei  = (const int*)d_in[1];
    const float* W1  = (const float*)d_in[2];
    const float* b1  = (const float*)d_in[3];
    const float* W2  = (const float*)d_in[4];
    const float* b2  = (const float*)d_in[5];
    float* out = (float*)d_out;

    const int*   esrc = ei;
    const int*   edst = ei + N_EDGES;

    // workspace carve-out
    char* ws = (char*)d_ws;
    size_t off = 0;
    auto carve = [&](size_t bytes) { char* p = ws + off; off = (off + bytes + 255) & ~(size_t)255; return p; };
    float*  dis  = (float*) carve((size_t)N_NODES * 4);                 // deg -> dis in place
    bf16_t* W1t  = (bf16_t*)carve((size_t)HID_DIM * IN_DIM * 2);
    bf16_t* W2t  = (bf16_t*)carve((size_t)64 * 64 * 2);
    float*  ht1  = (float*) carve((size_t)N_NODES * HID_DIM * 4);
    float*  agg1 = (float*) carve((size_t)N_NODES * HID_DIM * 4);
    bf16_t* h1bf = (bf16_t*)carve((size_t)N_NODES * HID_DIM * 2);
    float*  ht2  = (float*) carve((size_t)N_NODES * 64 * 4);
    float*  agg2 = (float*) carve((size_t)N_NODES * OUT_DIM * 4);
    (void)ws_size;

    hipMemsetAsync(dis,  0, (size_t)N_NODES * 4, stream);
    hipMemsetAsync(agg1, 0, (size_t)N_NODES * HID_DIM * 4, stream);
    hipMemsetAsync(agg2, 0, (size_t)N_NODES * OUT_DIM * 4, stream);

    k_deg<<<(N_EDGES + 255) / 256, 256, 0, stream>>>(edst, dis, N_EDGES);
    k_dis<<<(N_NODES + 255) / 256, 256, 0, stream>>>(dis, N_NODES);
    k_w1t<<<(IN_DIM * HID_DIM + 255) / 256, 256, 0, stream>>>(W1, W1t);
    k_w2t<<<(64 * 64 + 255) / 256, 256, 0, stream>>>(W2, W2t);

    const int mtiles = N_NODES / 16;            // 3125, exact
    const int gblocks = (mtiles + 7) / 8;       // 8 waves per block
    k_gemm_f32a<<<gblocks, 256, 0, stream>>>(x, W1t, ht1, N_NODES, IN_DIM);

    long long tot1 = (long long)N_EDGES * HID_DIM;
    k_scatter<<<(int)((tot1 + 255) / 256), 256, 0, stream>>>(esrc, edst, dis, ht1, agg1,
                                                             N_EDGES, HID_DIM, HID_DIM);
    k_post1<<<(N_NODES * HID_DIM + 255) / 256, 256, 0, stream>>>(agg1, ht1, dis, b1, h1bf, N_NODES);

    k_gemm_bf16a<<<gblocks, 256, 0, stream>>>(h1bf, W2t, ht2, N_NODES, HID_DIM);

    long long tot2 = (long long)N_EDGES * OUT_DIM;
    k_scatter<<<(int)((tot2 + 255) / 256), 256, 0, stream>>>(esrc, edst, dis, ht2, agg2,
                                                             N_EDGES, OUT_DIM, 64);
    k_post2<<<(N_NODES + 255) / 256, 256, 0, stream>>>(agg2, ht2, dis, b2, out, N_NODES);
}
